// UEFL_86964497809594
// MI455X (gfx1250) — matrix operand order repaired and checked by reading, not compiled, not run on hardware
//
#include <hip/hip_runtime.h>
#include <hip/hip_bf16.h>

typedef __attribute__((ext_vector_type(16))) _Float16 v16h;
typedef __attribute__((ext_vector_type(8)))  _Float16 v8h;
typedef __attribute__((ext_vector_type(8)))  float    v8f;

#define WMMA_F32_F16(a, b, c) \
  __builtin_amdgcn_wmma_f32_16x16x32_f16(false, (a), false, (b), (short)0, (c), false, false)

// ---------------------------------------------------------------------------
// Problem constants
// ---------------------------------------------------------------------------
#define BATCH   2048
#define DIM1    128     // conv1 out channels
#define DIM2    256     // conv2 out channels == embedding dim D
#define NPOS    64      // 8*8 spatial positions after encoder
#define MROWS   (BATCH * NPOS)   // 131072 rows of `flat`
#define KCONV2  2048    // 128*4*4
#define KCODES  1024    // max codebook rows (cb0 + cb1/cb2)
#define HIDDEN  512
#define CHFLAT  16384   // 256*64
#define OUTDIM  10

__device__ __forceinline__ const float* code_row(int j, int idx,
    const float* cb0, const float* cb1, const float* cb2) {
  if (j < 512) return cb0 + (size_t)j * DIM2;
  const float* cb = (idx == 1) ? cb1 : cb2;
  return cb + (size_t)(j - 512) * DIM2;
}

// Build a 16-element f16 fragment from two contiguous 16-byte LDS spans.
__device__ __forceinline__ v16h frag16(const _Float16* p0, const _Float16* p1) {
  v8h lo = *reinterpret_cast<const v8h*>(p0);
  v8h hi = *reinterpret_cast<const v8h*>(p1);
  return __builtin_shufflevector(lo, hi, 0,1,2,3,4,5,6,7,8,9,10,11,12,13,14,15);
}

// Convert 8 floats (two float4) to packed v8h.
__device__ __forceinline__ v8h cvt8(float4 a, float4 b) {
  v8h p;
  p[0]=(_Float16)a.x; p[1]=(_Float16)a.y; p[2]=(_Float16)a.z; p[3]=(_Float16)a.w;
  p[4]=(_Float16)b.x; p[5]=(_Float16)b.y; p[6]=(_Float16)b.z; p[7]=(_Float16)b.w;
  return p;
}

// ---------------------------------------------------------------------------
// conv1: (2048,3,32,32) -> relu -> h (2048,128,16,16) f16, stride2 k4 pad1
// ---------------------------------------------------------------------------
__global__ void conv1_relu_kernel(const float* __restrict__ x,
                                  const float* __restrict__ w1,
                                  const float* __restrict__ b1,
                                  _Float16* __restrict__ h) {
  int t = blockIdx.x * blockDim.x + threadIdx.x;   // exactly 2048*128*256
  int ow = t & 15;
  int oh = (t >> 4) & 15;
  int co = (t >> 8) & 127;
  int n  = t >> 15;
  float acc = b1[co];
  #pragma unroll
  for (int ci = 0; ci < 3; ++ci) {
    const float* xp = x + (((size_t)n * 3 + ci) << 10);   // 32*32
    const float* wp = w1 + ((co * 3 + ci) << 4);          // 4*4
    #pragma unroll
    for (int kh = 0; kh < 4; ++kh) {
      int ih = oh * 2 - 1 + kh;
      bool okh = (unsigned)ih < 32u;
      int ihc = ih & 31;
      #pragma unroll
      for (int kw = 0; kw < 4; ++kw) {
        int iw = ow * 2 - 1 + kw;
        bool ok = okh && ((unsigned)iw < 32u);
        float xv = xp[ihc * 32 + (iw & 31)];
        acc += (ok ? xv : 0.f) * wp[kh * 4 + kw];
      }
    }
  }
  h[t] = (_Float16)fmaxf(acc, 0.f);
}

// ---------------------------------------------------------------------------
// conv2 as implicit GEMM (WMMA): M=131072 (n,oh,ow), N=256 (cout), K=2048.
// 8 waves as 2x4 grid of 32x32 tiles -> block tile 64x128; 4 WMMA / K-step.
// Writes flat NHWC row-major: flat[(n*64+oh*8+ow)*256 + cout]
// ---------------------------------------------------------------------------
__global__ void conv2_wmma_kernel(const _Float16* __restrict__ h,
                                  const float* __restrict__ w2,
                                  const float* __restrict__ b2,
                                  float* __restrict__ flat) {
  __shared__ _Float16 As[64][40];    // 64 M x 32 K  (+pad; row = 80B, 16B-aligned)
  __shared__ _Float16 Bt[128][40];   // 128 N x 32 K (transposed for vector frags)
  const int tid  = threadIdx.x;
  const int lane = tid & 31, wave = tid >> 5;
  const int wr = wave >> 2, wcol = wave & 3;
  const int half = lane >> 4, l16 = lane & 15;
  const int blockM = blockIdx.x * 64;
  const int blockN = blockIdx.y * 128;
  v8f c00 = {}, c01 = {}, c10 = {}, c11 = {};

  // A-stage thread mapping (fixed across K loop)
  const int ar = tid >> 2;
  const int acb = (tid & 3) * 8;
  const int am = blockM + ar;
  const int img = am >> 6, pos = am & 63;
  const int ohh = pos >> 3, oww = pos & 7;
  const _Float16* hbase = h + ((size_t)img * DIM1) * 256;  // [ci][ih][iw]

  for (int k0 = 0; k0 < KCONV2; k0 += 32) {
    { // stage A: im2col gather, clamped loads + select (no divergence)
      #pragma unroll
      for (int j = 0; j < 8; ++j) {
        int k = k0 + acb + j;
        int ci = k >> 4, rem = k & 15;
        int kh = rem >> 2, kw = rem & 3;
        int ih = ohh * 2 - 1 + kh, iw = oww * 2 - 1 + kw;
        bool ok = ((unsigned)ih < 16u) && ((unsigned)iw < 16u);
        _Float16 vv = hbase[(size_t)ci * 256 + (ih & 15) * 16 + (iw & 15)];
        As[ar][acb + j] = ok ? vv : (_Float16)0.f;
      }
    }
    { // stage B transposed: Bt[n][k] = w2[n][k0+k]; contiguous float4 loads
      int n = tid >> 1;                  // 0..127
      int kb = (tid & 1) * 16;
      const float4* src = reinterpret_cast<const float4*>(
          w2 + (size_t)(blockN + n) * KCONV2 + k0 + kb);
      float4 x0 = src[0], x1 = src[1], x2 = src[2], x3 = src[3];
      *reinterpret_cast<v8h*>(&Bt[n][kb])     = cvt8(x0, x1);
      *reinterpret_cast<v8h*>(&Bt[n][kb + 8]) = cvt8(x2, x3);
    }
    __syncthreads();
    const _Float16* ar0 = &As[wr * 32 + l16][0];
    const _Float16* ar1 = &As[wr * 32 + 16 + l16][0];
    const _Float16* br0 = &Bt[wcol * 32 + l16][0];
    const _Float16* br1 = &Bt[wcol * 32 + 16 + l16][0];
    v16h a0 = frag16(ar0 + half * 8, ar0 + 16 + half * 8);
    v16h a1 = frag16(ar1 + half * 8, ar1 + 16 + half * 8);
    v16h b0 = frag16(br0 + half * 8, br0 + 16 + half * 8);
    v16h b1 = frag16(br1 + half * 8, br1 + 16 + half * 8);
    c00 = WMMA_F32_F16(a0, b0, c00);
    c01 = WMMA_F32_F16(a0, b1, c01);
    c10 = WMMA_F32_F16(a1, b0, c10);
    c11 = WMMA_F32_F16(a1, b1, c11);
    __syncthreads();
  }
  // epilogue: bias + relu
  #pragma unroll
  for (int v = 0; v < 8; ++v) {
    int m0 = blockM + wr * 32 + v + half * 8;
    int n0 = blockN + wcol * 32 + l16;
    flat[(size_t)m0 * DIM2 + n0]            = fmaxf(c00[v] + b2[n0], 0.f);
    flat[(size_t)m0 * DIM2 + n0 + 16]       = fmaxf(c01[v] + b2[n0 + 16], 0.f);
    flat[(size_t)(m0 + 16) * DIM2 + n0]     = fmaxf(c10[v] + b2[n0], 0.f);
    flat[(size_t)(m0 + 16) * DIM2 + n0 + 16]= fmaxf(c11[v] + b2[n0 + 16], 0.f);
  }
}

// ---------------------------------------------------------------------------
// Row / code squared norms (float4 loops)
// ---------------------------------------------------------------------------
__global__ void row_sq_kernel(const float* __restrict__ flat, float* __restrict__ fsq) {
  int m = blockIdx.x * blockDim.x + threadIdx.x;
  if (m >= MROWS) return;
  const float4* r = reinterpret_cast<const float4*>(flat + (size_t)m * DIM2);
  float s = 0.f;
  for (int d = 0; d < DIM2 / 4; ++d) {
    float4 v = r[d];
    s += v.x * v.x + v.y * v.y + v.z * v.z + v.w * v.w;
  }
  fsq[m] = s;
}

__global__ void code_sq_kernel(const int* __restrict__ idxp,
                               const float* cb0, const float* cb1, const float* cb2,
                               float* __restrict__ csq) {
  int j = blockIdx.x * blockDim.x + threadIdx.x;
  if (j >= KCODES) return;
  int idx = *idxp;
  int Kc = (idx == 0) ? 512 : 1024;
  float s = 1e30f;
  if (j < Kc) {
    const float4* r = reinterpret_cast<const float4*>(code_row(j, idx, cb0, cb1, cb2));
    s = 0.f;
    for (int d = 0; d < DIM2 / 4; ++d) {
      float4 v = r[d];
      s += v.x * v.x + v.y * v.y + v.z * v.z + v.w * v.w;
    }
  }
  csq[j] = s;
}

// ---------------------------------------------------------------------------
// Fused distance GEMM + argmin (WMMA). 4x2 wave grid of 16x16 tiles.
// A (64x256) staged once; codes streamed 32 at a time; dist epilogue + argmin.
// ---------------------------------------------------------------------------
__global__ void dist_argmin_wmma_kernel(const float* __restrict__ flat,
                                        const float* __restrict__ fsq,
                                        const float* __restrict__ csq,
                                        const int* __restrict__ idxp,
                                        const float* cb0, const float* cb1, const float* cb2,
                                        int* __restrict__ enc_idx) {
  __shared__ _Float16 Af[64][264];   // row = 528B, 16B-aligned
  __shared__ _Float16 Bs[32][264];   // 32 codes x 256 dims (code-major == B^T)
  const int tid  = threadIdx.x;
  const int lane = tid & 31, wave = tid >> 5;
  const int wr = wave >> 1, wc = wave & 1;
  const int half = lane >> 4, l16 = lane & 15;
  const int blockM = blockIdx.x * 64;
  const int idx = *idxp;
  const int Kc = (idx == 0) ? 512 : 1024;

  { // stage Af with float4 loads: 64 elems/thread
    int r = tid >> 2;
    int cb = (tid & 3) * 64;
    const float4* src = reinterpret_cast<const float4*>(
        flat + (size_t)(blockM + r) * DIM2 + cb);
    #pragma unroll
    for (int j = 0; j < 8; ++j) {
      float4 x0 = src[2 * j], x1 = src[2 * j + 1];
      *reinterpret_cast<v8h*>(&Af[r][cb + 8 * j]) = cvt8(x0, x1);
    }
  }

  float fsqv[8];
  #pragma unroll
  for (int v = 0; v < 8; ++v) fsqv[v] = fsq[blockM + wr * 16 + v + half * 8];
  float bestV[8]; int bestI[8];
  #pragma unroll
  for (int v = 0; v < 8; ++v) { bestV[v] = 3.4e38f; bestI[v] = 0; }
  __syncthreads();

  for (int nc = 0; nc < Kc; nc += 32) {
    { // stage 32 code rows, float4 loads: 32 elems/thread
      int r = tid >> 3;
      int cb = (tid & 7) * 32;
      const float4* src = reinterpret_cast<const float4*>(
          code_row(nc + r, idx, cb0, cb1, cb2) + cb);
      #pragma unroll
      for (int j = 0; j < 4; ++j) {
        float4 x0 = src[2 * j], x1 = src[2 * j + 1];
        *reinterpret_cast<v8h*>(&Bs[r][cb + 8 * j]) = cvt8(x0, x1);
      }
    }
    __syncthreads();
    v8f acc = {};
    const _Float16* arow = &Af[wr * 16 + l16][0];
    const _Float16* brow = &Bs[wc * 16 + l16][0];
    #pragma unroll
    for (int k0 = 0; k0 < DIM2; k0 += 32) {
      v16h a = frag16(arow + k0 + half * 8, arow + k0 + 16 + half * 8);
      v16h b = frag16(brow + k0 + half * 8, brow + k0 + 16 + half * 8);
      acc = WMMA_F32_F16(a, b, acc);
    }
    int j = nc + wc * 16 + l16;
    float cs = csq[j];
    #pragma unroll
    for (int v = 0; v < 8; ++v) {
      float d = fsqv[v] + cs - 2.f * acc[v];
      if (d < bestV[v]) { bestV[v] = d; bestI[v] = j; }
    }
    __syncthreads();
  }

  // cross-lane reduction: reuse Bs storage (16,896 B >= 16 KB needed)
  float* redV = reinterpret_cast<float*>(&Bs[0][0]);   // [64][32]
  int*   redI = reinterpret_cast<int*>(redV + 64 * 32);
  #pragma unroll
  for (int v = 0; v < 8; ++v) {
    int row = wr * 16 + v + half * 8;
    int slot = wc * 16 + l16;
    redV[row * 32 + slot] = bestV[v];
    redI[row * 32 + slot] = bestI[v];
  }
  __syncthreads();
  if (tid < 64) {
    float bv = redV[tid * 32]; int bi = redI[tid * 32];
    for (int s = 1; s < 32; ++s) {
      float v = redV[tid * 32 + s];
      if (v < bv) { bv = v; bi = redI[tid * 32 + s]; }
    }
    enc_idx[blockM + tid] = bi;
  }
}

// ---------------------------------------------------------------------------
// Loss (1.25 * mse(quantized, flat)) partials + code histogram
// ---------------------------------------------------------------------------
__global__ void loss_hist_kernel(const float* __restrict__ flat,
                                 const int* __restrict__ enc_idx,
                                 const int* __restrict__ idxp,
                                 const float* cb0, const float* cb1, const float* cb2,
                                 float* __restrict__ lossSum,
                                 float* __restrict__ hist) {
  int m = blockIdx.x * blockDim.x + threadIdx.x;
  if (m >= MROWS) return;
  int idx = *idxp;
  int e = enc_idx[m];
  const float4* cr = reinterpret_cast<const float4*>(code_row(e, idx, cb0, cb1, cb2));
  const float4* fr = reinterpret_cast<const float4*>(flat + (size_t)m * DIM2);
  float s = 0.f;
  for (int d = 0; d < DIM2 / 4; ++d) {
    float4 c = cr[d], f = fr[d];
    float dx = c.x - f.x, dy = c.y - f.y, dz = c.z - f.z, dw = c.w - f.w;
    s += dx * dx + dy * dy + dz * dz + dw * dw;
  }
  atomicAdd(lossSum, s);
  atomicAdd(&hist[e], 1.f);
}

// ---------------------------------------------------------------------------
// fc1 + exact GELU (WMMA): M=2048, N=512, K=16384. 2x4 wave grid, 32x32 tiles.
// A'[b][pos*256+co] = codes[enc_idx[b*64+pos]][co]
// B'[k'][n] = fc1_w[n][co*64+pos]
// ---------------------------------------------------------------------------
__global__ void fc1_gelu_wmma_kernel(const int* __restrict__ enc_idx,
                                     const int* __restrict__ idxp,
                                     const float* cb0, const float* cb1, const float* cb2,
                                     const float* __restrict__ fc1_w,
                                     const float* __restrict__ fc1_b,
                                     float* __restrict__ h1) {
  __shared__ _Float16 As[64][40];
  __shared__ _Float16 Bt[128][40];   // transposed: [n][k]
  const int tid  = threadIdx.x;
  const int lane = tid & 31, wave = tid >> 5;
  const int wr = wave >> 2, wcol = wave & 3;
  const int half = lane >> 4, l16 = lane & 15;
  const int blockM = blockIdx.x * 64;   // batch rows
  const int blockN = blockIdx.y * 128;  // hidden units
  const int idx = *idxp;
  v8f c00 = {}, c01 = {}, c10 = {}, c11 = {};

  for (int k0 = 0; k0 < CHFLAT; k0 += 32) {
    const int pos = k0 >> 8;        // constant within 32-wide chunk
    const int co0 = k0 & 255;
    { // stage A: codebook gather; code rows contiguous in co -> float4 loads
      int r = tid >> 2;
      int cbase = (tid & 3) * 8;
      int m = blockM + r;
      int e = enc_idx[m * NPOS + pos];
      const float4* cr = reinterpret_cast<const float4*>(
          code_row(e, idx, cb0, cb1, cb2) + co0 + cbase);
      *reinterpret_cast<v8h*>(&As[r][cbase]) = cvt8(cr[0], cr[1]);
    }
    { // stage B transposed: strided gather of fc1_w
      int n = tid >> 1;                 // 0..127
      int kb = (tid & 1) * 16;
      const float* src = fc1_w + (size_t)(blockN + n) * CHFLAT + pos
                       + (size_t)(co0 + kb) * NPOS;
      v8h p0, p1;
      #pragma unroll
      for (int j = 0; j < 8; ++j) p0[j] = (_Float16)src[(size_t)j * NPOS];
      #pragma unroll
      for (int j = 0; j < 8; ++j) p1[j] = (_Float16)src[(size_t)(8 + j) * NPOS];
      *reinterpret_cast<v8h*>(&Bt[n][kb])     = p0;
      *reinterpret_cast<v8h*>(&Bt[n][kb + 8]) = p1;
    }
    __syncthreads();
    const _Float16* ar0 = &As[wr * 32 + l16][0];
    const _Float16* ar1 = &As[wr * 32 + 16 + l16][0];
    const _Float16* br0 = &Bt[wcol * 32 + l16][0];
    const _Float16* br1 = &Bt[wcol * 32 + 16 + l16][0];
    v16h a0 = frag16(ar0 + half * 8, ar0 + 16 + half * 8);
    v16h a1 = frag16(ar1 + half * 8, ar1 + 16 + half * 8);
    v16h b0 = frag16(br0 + half * 8, br0 + 16 + half * 8);
    v16h b1 = frag16(br1 + half * 8, br1 + 16 + half * 8);
    c00 = WMMA_F32_F16(a0, b0, c00);
    c01 = WMMA_F32_F16(a0, b1, c01);
    c10 = WMMA_F32_F16(a1, b0, c10);
    c11 = WMMA_F32_F16(a1, b1, c11);
    __syncthreads();
  }
  #pragma unroll
  for (int v = 0; v < 8; ++v) {
    int m0 = blockM + wr * 32 + v + half * 8;
    int n0 = blockN + wcol * 32 + l16;
    float x00 = c00[v] + fc1_b[n0];
    float x01 = c01[v] + fc1_b[n0 + 16];
    float x10 = c10[v] + fc1_b[n0];
    float x11 = c11[v] + fc1_b[n0 + 16];
    h1[(size_t)m0 * HIDDEN + n0]        = 0.5f * x00 * (1.f + erff(x00 * 0.70710678f));
    h1[(size_t)m0 * HIDDEN + n0 + 16]   = 0.5f * x01 * (1.f + erff(x01 * 0.70710678f));
    h1[(size_t)(m0+16) * HIDDEN + n0]      = 0.5f * x10 * (1.f + erff(x10 * 0.70710678f));
    h1[(size_t)(m0+16) * HIDDEN + n0 + 16] = 0.5f * x11 * (1.f + erff(x11 * 0.70710678f));
  }
}

// ---------------------------------------------------------------------------
// fc2 (2048x512 @ 512x10) — tiny, plain VALU
// ---------------------------------------------------------------------------
__global__ void fc2_kernel(const float* __restrict__ h1,
                           const float* __restrict__ fc2_w,
                           const float* __restrict__ fc2_b,
                           float* __restrict__ out) {
  int t = blockIdx.x * blockDim.x + threadIdx.x;
  if (t >= BATCH * OUTDIM) return;
  int o = t % OUTDIM, b = t / OUTDIM;
  const float4* hr = reinterpret_cast<const float4*>(h1 + (size_t)b * HIDDEN);
  const float4* wr = reinterpret_cast<const float4*>(fc2_w + (size_t)o * HIDDEN);
  float s = fc2_b[o];
  for (int k = 0; k < HIDDEN / 4; ++k) {
    float4 a = hr[k], w = wr[k];
    s += a.x * w.x + a.y * w.y + a.z * w.z + a.w * w.w;
  }
  out[t] = s;
}

__global__ void zero_kernel(float* p, int n) {
  int t = blockIdx.x * blockDim.x + threadIdx.x;
  if (t < n) p[t] = 0.f;
}

__global__ void finalize_kernel(const float* __restrict__ lossSum,
                                const float* __restrict__ hist,
                                const int* __restrict__ idxp,
                                float* __restrict__ out) {
  __shared__ float sh[1024];
  int t = threadIdx.x;
  int idx = *idxp;
  int Kc = (idx == 0) ? 512 : 1024;
  float s = 0.f;
  if (t < Kc) {
    float p = hist[t] * (1.f / (float)MROWS);
    s = p * logf(p + 1e-10f);
  }
  sh[t] = s;
  __syncthreads();
  for (int st = 512; st > 0; st >>= 1) {
    if (t < st) sh[t] += sh[t + st];
    __syncthreads();
  }
  if (t == 0) {
    out[BATCH * OUTDIM]     = 1.25f * lossSum[0] / ((float)MROWS * (float)DIM2);
    out[BATCH * OUTDIM + 1] = expf(-sh[0]);
  }
}

// ---------------------------------------------------------------------------
extern "C" void kernel_launch(void* const* d_in, const int* in_sizes, int n_in,
                              void* d_out, int out_size, void* d_ws, size_t ws_size,
                              hipStream_t stream) {
  const float* x     = (const float*)d_in[0];
  const int*   idxp  = (const int*)d_in[1];
  const float* w1    = (const float*)d_in[2];
  const float* b1    = (const float*)d_in[3];
  const float* w2    = (const float*)d_in[4];
  const float* b2    = (const float*)d_in[5];
  const float* cb0   = (const float*)d_in[6];
  const float* cb1   = (const float*)d_in[7];
  const float* cb2   = (const float*)d_in[8];
  const float* fc1_w = (const float*)d_in[9];
  const float* fc1_b = (const float*)d_in[10];
  const float* fc2_w = (const float*)d_in[11];
  const float* fc2_b = (const float*)d_in[12];
  float* out = (float*)d_out;

  // workspace layout (~274 MB)
  char* ws = (char*)d_ws;
  _Float16* h    = (_Float16*)(ws);                 // 2048*128*16*16 f16 = 134,217,728
  float* flat    = (float*)(ws + 134217728);        // 131072*256 f32     = 134,217,728
  float* h1      = (float*)(ws + 268435456);        // 2048*512 f32       =   4,194,304
  int*   enc_idx = (int*)(ws + 272629760);          // 131072 i32         =     524,288
  float* fsq     = (float*)(ws + 273154048);        // 131072 f32         =     524,288
  float* csq     = (float*)(ws + 273678336);        // 1024 f32
  float* hist    = (float*)(ws + 273682432);        // 1024 f32
  float* lossSum = (float*)(ws + 273686528);        // 1 f32

  zero_kernel<<<4, 256, 0, stream>>>(hist, 1024);
  zero_kernel<<<1, 32, 0, stream>>>(lossSum, 1);

  conv1_relu_kernel<<<(BATCH * DIM1 * 256) / 256, 256, 0, stream>>>(x, w1, b1, h);
  conv2_wmma_kernel<<<dim3(MROWS / 64, DIM2 / 128), 256, 0, stream>>>(h, w2, b2, flat);
  row_sq_kernel<<<MROWS / 256, 256, 0, stream>>>(flat, fsq);
  code_sq_kernel<<<KCODES / 256, 256, 0, stream>>>(idxp, cb0, cb1, cb2, csq);
  dist_argmin_wmma_kernel<<<MROWS / 64, 256, 0, stream>>>(flat, fsq, csq, idxp,
                                                          cb0, cb1, cb2, enc_idx);
  loss_hist_kernel<<<MROWS / 256, 256, 0, stream>>>(flat, enc_idx, idxp,
                                                    cb0, cb1, cb2, lossSum, hist);
  fc1_gelu_wmma_kernel<<<dim3(BATCH / 64, HIDDEN / 128), 256, 0, stream>>>(
      enc_idx, idxp, cb0, cb1, cb2, fc1_w, fc1_b, h1);
  fc2_kernel<<<(BATCH * OUTDIM + 255) / 256, 256, 0, stream>>>(h1, fc2_w, fc2_b, out);
  finalize_kernel<<<1, 1024, 0, stream>>>(lossSum, hist, idxp, out);
}
